// MultiHeadSelfAttention_43739946942484
// MI455X (gfx1250) — compile-verified, matrix operand order checked by reading, not run on hardware
//
#include <hip/hip_runtime.h>
#include <hip/hip_bf16.h>
#include <math.h>

// ---------------- problem constants ----------------
constexpr int kE = 1024;   // embed
constexpr int kH = 16;     // heads
constexpr int kD = 64;     // head dim
constexpr int kS = 2048;   // seq len
constexpr int kB = 4;      // batch
constexpr int kTok = kB * kS;          // 8192 tokens

typedef __attribute__((ext_vector_type(16))) __bf16 v16bf;
typedef __attribute__((ext_vector_type(8)))  float  v8f;
typedef __attribute__((ext_vector_type(4)))  int    v4i;

// ---------------- async global->LDS copy (CDNA5 GLOBAL_LOAD_ASYNC_TO_LDS_B128) ----
#if __has_builtin(__builtin_amdgcn_global_load_async_to_lds_b128)
#define ASYNC_COPY 1
#else
#define ASYNC_COPY 0
#endif

#if ASYNC_COPY
typedef __attribute__((address_space(1))) v4i gv4i;   // global int4
typedef __attribute__((address_space(3))) v4i lv4i;   // LDS int4
#endif

__device__ __forceinline__ void cp_b128(void* lds, const void* g) {
#if ASYNC_COPY
  // generic->AS casts via integers: AS3 pointers are the low 32 bits of the
  // generic address.
  __builtin_amdgcn_global_load_async_to_lds_b128(
      (gv4i*)(unsigned long long)g,
      (lv4i*)(unsigned int)(unsigned long long)lds,
      0, 0);
#else
  *(uint4*)lds = *(const uint4*)g;
#endif
}

__device__ __forceinline__ void wait_async() {
#if ASYNC_COPY
#if __has_builtin(__builtin_amdgcn_s_wait_asynccnt)
  __builtin_amdgcn_s_wait_asynccnt(0);
#else
  asm volatile("s_wait_asynccnt 0x0" ::: "memory");
#endif
#endif
}

// round-to-nearest-even fp32 -> bf16
__device__ __forceinline__ __bf16 f2bf(float f) {
  unsigned u = __float_as_uint(f);
  u += 0x7FFFu + ((u >> 16) & 1u);
  unsigned short h = (unsigned short)(u >> 16);
  __bf16 r;
  __builtin_memcpy(&r, &h, sizeof(r));
  return r;
}

// ---------------- fragment loader (ISA 7.12.2 layout) ----------------
// A (16x32, MxK): lane<16 -> row M=lane, K in {0..7} and {16..23};
//                 lane>=16 -> row M=lane-16, K in {8..15} and {24..31}.
// B (32x16, KxN) stored N-major in LDS: same pattern with N=lane&15.
__device__ __forceinline__ v16bf load_frag(const __bf16* p, int stride, int k0, int lane) {
  const int r  = lane & 15;
  const int kh = (lane & 16) ? 8 : 0;
  union { v16bf v; uint4 q[2]; } f;
  const __bf16* rp = p + r * stride + k0 + kh;
  f.q[0] = *(const uint4*)(rp);
  f.q[1] = *(const uint4*)(rp + 16);
  return f.v;
}

__device__ __forceinline__ v8f wmma_bf16(v16bf a, v16bf b, v8f c) {
  return __builtin_amdgcn_wmma_f32_16x16x32_bf16(
      false, a, false, b, (short)0, c, false, false);
}

// ---------------- fp32 -> bf16 conversion ----------------
__global__ __launch_bounds__(256) void cvt_f32_bf16(
    const float* __restrict__ in, __bf16* __restrict__ out, int n) {
  int i = blockIdx.x * blockDim.x + threadIdx.x;
  if (i < n) out[i] = f2bf(in[i]);
}

// ---------------- tiled WMMA GEMM: C[M,1024] = A[M,1024] @ W^T ----------------
// 128x128 block tile, 8 waves each computing 32x64 (2x4 16x16 tiles).
// Double-buffered LDS, async global->LDS staging.
// mode 0: bf16 out -> [B,H,S,D] (Q,K; scale folded)
// mode 1: bf16 out -> [B,H,D,S] (V transposed)
// mode 2: f32 out [M,1024] + bias (final projection)
__global__ __launch_bounds__(256)
void gemm_bf16(const __bf16* __restrict__ A,
               const __bf16* __restrict__ W,
               __bf16* __restrict__ outB,
               float* __restrict__ outF,
               const float* __restrict__ bias,
               float scale, int mode) {
  __shared__ __align__(16) __bf16 sA[2][128][32];
  __shared__ __align__(16) __bf16 sB[2][128][32];   // N-major: sB[buf][n][k]

  const int tid  = threadIdx.x;
  const int lane = tid & 31;
  const int wave = tid >> 5;
  const int m0 = blockIdx.y * 128;
  const int n0 = blockIdx.x * 128;
  const int wr = wave >> 1;      // 0..3 -> 32-row strip
  const int wc = wave & 1;       // 0..1 -> 64-col strip

  v8f acc[2][4];
#pragma unroll
  for (int i = 0; i < 2; ++i)
#pragma unroll
    for (int j = 0; j < 4; ++j) acc[i][j] = (v8f){};

  // stage one 128x32 K-slice of A and B into buffer b (1024 b128 chunks, 4/thread)
  auto stage = [&](int k0, int b) {
#pragma unroll
    for (int i = 0; i < 2; ++i) {
      int idx = tid + i * 256;
      int row = idx >> 2, ch = (idx & 3) * 8;
      cp_b128(&sA[b][row][ch], A + (size_t)(m0 + row) * kE + k0 + ch);
      cp_b128(&sB[b][row][ch], W + (size_t)(n0 + row) * kE + k0 + ch);
    }
  };

  stage(0, 0);
  wait_async();
  __syncthreads();

  const int KT = kE / 32;
  for (int kt = 0; kt < KT; ++kt) {
    const int cur = kt & 1;
    if (kt + 1 < KT) stage((kt + 1) * 32, cur ^ 1);   // prefetch next slice

    v16bf a0 = load_frag(&sA[cur][wr * 32][0],      32, 0, lane);
    v16bf a1 = load_frag(&sA[cur][wr * 32 + 16][0], 32, 0, lane);
    v16bf b0 = load_frag(&sB[cur][wc * 64][0],      32, 0, lane);
    v16bf b1 = load_frag(&sB[cur][wc * 64 + 16][0], 32, 0, lane);
    v16bf b2 = load_frag(&sB[cur][wc * 64 + 32][0], 32, 0, lane);
    v16bf b3 = load_frag(&sB[cur][wc * 64 + 48][0], 32, 0, lane);

    acc[0][0] = wmma_bf16(a0, b0, acc[0][0]);
    acc[0][1] = wmma_bf16(a0, b1, acc[0][1]);
    acc[0][2] = wmma_bf16(a0, b2, acc[0][2]);
    acc[0][3] = wmma_bf16(a0, b3, acc[0][3]);
    acc[1][0] = wmma_bf16(a1, b0, acc[1][0]);
    acc[1][1] = wmma_bf16(a1, b1, acc[1][1]);
    acc[1][2] = wmma_bf16(a1, b2, acc[1][2]);
    acc[1][3] = wmma_bf16(a1, b3, acc[1][3]);

    wait_async();      // next-slice copies landed
    __syncthreads();   // everyone done reading cur
  }

  // epilogue: C layout -> m = r + 8*(lane>=16), n = lane&15
  const int n_in = lane & 15;
  const int mhi  = (lane & 16) ? 8 : 0;
#pragma unroll
  for (int i = 0; i < 2; ++i)
#pragma unroll
    for (int j = 0; j < 4; ++j) {
      int gn = n0 + wc * 64 + j * 16 + n_in;
      float bv = (mode == 2) ? bias[gn] : 0.0f;
#pragma unroll
      for (int r = 0; r < 8; ++r) {
        int gm = m0 + wr * 32 + i * 16 + r + mhi;
        float v = acc[i][j][r];
        if (mode == 2) {
          outF[(size_t)gm * kE + gn] = v + bv;
        } else {
          int b = gm >> 11, s = gm & (kS - 1);
          int h = gn >> 6,  d = gn & (kD - 1);
          size_t idx = (mode == 0)
              ? ((((size_t)b * kH + h) * kS + s) * kD + d)    // [B,H,S,D]
              : ((((size_t)b * kH + h) * kD + d) * kS + s);   // [B,H,D,S]
          outB[idx] = f2bf(v * scale);
        }
      }
    }
}

// ---------------- flash attention over one (b,h, 64-query) block ----------------
// Double-buffered K tiles (prefetched one block ahead), V copy issued at top of
// iteration and waited only before P@V (hidden behind scores + softmax).
__global__ __launch_bounds__(256)
void attn_fa(const __bf16* __restrict__ Q,   // [B,H,S,D]
             const __bf16* __restrict__ K,   // [B,H,S,D]
             const __bf16* __restrict__ VT,  // [B,H,D,S]
             __bf16* __restrict__ O) {       // [B,S,H,D] == [Tok, E]
  __shared__ __align__(16) __bf16 sQ[64][64];
  __shared__ __align__(16) __bf16 sK[2][64][64];   // [key][d], N-major for scores
  __shared__ __align__(16) __bf16 sVT[64][64];     // [d][key], N-major for P@V
  __shared__ __align__(16) float  sS[64][64];
  __shared__ __align__(16) __bf16 sP[64][64];
  __shared__ float m_s[64], l_s[64], a_s[64];

  const int tid  = threadIdx.x;
  const int lane = tid & 31;
  const int wave = tid >> 5;
  const int qb = blockIdx.x;
  const int bh = blockIdx.y;

  const __bf16* Qp = Q  + (size_t)bh * kS * kD;
  const __bf16* Kp = K  + (size_t)bh * kS * kD;
  const __bf16* Vp = VT + (size_t)bh * kD * kS;

  auto stageK = [&](int kb, int b) {
#pragma unroll
    for (int i = 0; i < 2; ++i) {
      int idx = tid + i * 256;
      int row = idx >> 3, ch = (idx & 7) * 8;
      cp_b128(&sK[b][row][ch], Kp + (size_t)(kb * 64 + row) * kD + ch);
    }
  };
  auto stageV = [&](int kb) {
#pragma unroll
    for (int i = 0; i < 2; ++i) {
      int idx = tid + i * 256;
      int row = idx >> 3, ch = (idx & 7) * 8;
      cp_b128(&sVT[row][ch], Vp + (size_t)row * kS + kb * 64 + ch);
    }
  };

  // stage Q tile + first K tile
#pragma unroll
  for (int i = 0; i < 2; ++i) {
    int idx = tid + i * 256;
    int row = idx >> 3, ch = (idx & 7) * 8;
    cp_b128(&sQ[row][ch], Qp + (size_t)(qb * 64 + row) * kD + ch);
  }
  stageK(0, 0);
  if (tid < 64) { m_s[tid] = -1e30f; l_s[tid] = 0.0f; }
  wait_async();
  __syncthreads();

  const int mt  = wave >> 1;
  const int nt0 = (wave & 1) * 2;
  const int n_in = lane & 15;
  const int mhi  = (lane & 16) ? 8 : 0;

  v8f o0 = {}, o1 = {};

  const int NB = kS / 64;
  for (int kb = 0; kb < NB; ++kb) {
    const int cur = kb & 1;
    stageV(kb);                                // V for this block (hidden by softmax)
    if (kb + 1 < NB) stageK(kb + 1, cur ^ 1);  // prefetch next K

    // scores S = Q @ K^T (scale already folded into Q and K)
    v8f s0 = {}, s1 = {};
#pragma unroll
    for (int d0 = 0; d0 < kD; d0 += 32) {
      v16bf a  = load_frag(&sQ[mt * 16][0], 64, d0, lane);
      v16bf b0 = load_frag(&sK[cur][nt0 * 16][0], 64, d0, lane);
      v16bf b1 = load_frag(&sK[cur][(nt0 + 1) * 16][0], 64, d0, lane);
      s0 = wmma_bf16(a, b0, s0);
      s1 = wmma_bf16(a, b1, s1);
    }
#pragma unroll
    for (int r = 0; r < 8; ++r) {
      sS[mt * 16 + r + mhi][nt0 * 16 + n_in]       = s0[r];
      sS[mt * 16 + r + mhi][(nt0 + 1) * 16 + n_in] = s1[r];
    }
    __syncthreads();

    // online softmax: 4 threads per row, shfl_xor over the 4 partials
    {
      int row = tid >> 2, c0 = (tid & 3) * 16;
      float mx = -1e30f;
#pragma unroll
      for (int c = 0; c < 16; ++c) mx = fmaxf(mx, sS[row][c0 + c]);
      mx = fmaxf(mx, __shfl_xor(mx, 1, 32));
      mx = fmaxf(mx, __shfl_xor(mx, 2, 32));
      float mold = m_s[row];
      float mnew = fmaxf(mold, mx);
      float alpha = __expf(mold - mnew);
      float psum = 0.0f;
#pragma unroll
      for (int c = 0; c < 16; ++c) {
        float p = __expf(sS[row][c0 + c] - mnew);
        sP[row][c0 + c] = f2bf(p);
        psum += p;
      }
      psum += __shfl_xor(psum, 1, 32);
      psum += __shfl_xor(psum, 2, 32);
      if ((tid & 3) == 0) {
        m_s[row] = mnew;
        a_s[row] = alpha;
        l_s[row] = l_s[row] * alpha + psum;
      }
    }
    __syncthreads();

    // rescale O
#pragma unroll
    for (int r = 0; r < 8; ++r) {
      float al = a_s[mt * 16 + r + mhi];
      o0[r] *= al;
      o1[r] *= al;
    }

    wait_async();      // V (and next K) landed
    __syncthreads();

    // O += P @ V
#pragma unroll
    for (int k0 = 0; k0 < 64; k0 += 32) {
      v16bf a  = load_frag(&sP[mt * 16][0], 64, k0, lane);
      v16bf b0 = load_frag(&sVT[nt0 * 16][0], 64, k0, lane);
      v16bf b1 = load_frag(&sVT[(nt0 + 1) * 16][0], 64, k0, lane);
      o0 = wmma_bf16(a, b0, o0);
      o1 = wmma_bf16(a, b1, o1);
    }
    __syncthreads();   // protect sVT/sK/sS/sP for next iteration
  }

  // normalize and store O as [B,S,H,D]
  const int b = bh >> 4, h = bh & 15;
#pragma unroll
  for (int r = 0; r < 8; ++r) {
    int lrow = mt * 16 + r + mhi;
    float inv = 1.0f / l_s[lrow];
    int srow = qb * 64 + lrow;
    size_t base = (((size_t)b * kS + srow) * kH + h) * kD;
    O[base + nt0 * 16 + n_in]       = f2bf(o0[r] * inv);
    O[base + (nt0 + 1) * 16 + n_in] = f2bf(o1[r] * inv);
  }
}

// ---------------- host-side orchestration ----------------
extern "C" void kernel_launch(void* const* d_in, const int* in_sizes, int n_in,
                              void* d_out, int out_size, void* d_ws, size_t ws_size,
                              hipStream_t stream) {
  (void)in_sizes; (void)n_in; (void)out_size; (void)ws_size;
  const float* x  = (const float*)d_in[0];
  const float* Wq = (const float*)d_in[1];
  const float* Wk = (const float*)d_in[2];
  const float* Wv = (const float*)d_in[3];
  const float* Wo = (const float*)d_in[4];
  const float* bo = (const float*)d_in[5];
  float* out = (float*)d_out;

  char* ws = (char*)d_ws;
  size_t off = 0;
  auto carve = [&](size_t bytes) { char* p = ws + off; off += bytes; return p; };
  __bf16* xb  = (__bf16*)carve((size_t)kTok * kE * 2);
  __bf16* wqb = (__bf16*)carve((size_t)kE * kE * 2);
  __bf16* wkb = (__bf16*)carve((size_t)kE * kE * 2);
  __bf16* wvb = (__bf16*)carve((size_t)kE * kE * 2);
  __bf16* wob = (__bf16*)carve((size_t)kE * kE * 2);
  __bf16* Qb  = (__bf16*)carve((size_t)kTok * kE * 2);   // [B,H,S,D]
  __bf16* Kb  = (__bf16*)carve((size_t)kTok * kE * 2);   // [B,H,S,D]
  __bf16* VTb = (__bf16*)carve((size_t)kTok * kE * 2);   // [B,H,D,S]
  __bf16* Ob  = (__bf16*)carve((size_t)kTok * kE * 2);   // [B,S,H,D]

  const int nX = kTok * kE;
  const int nW = kE * kE;
  cvt_f32_bf16<<<nX / 256, 256, 0, stream>>>(x,  xb,  nX);
  cvt_f32_bf16<<<nW / 256, 256, 0, stream>>>(Wq, wqb, nW);
  cvt_f32_bf16<<<nW / 256, 256, 0, stream>>>(Wk, wkb, nW);
  cvt_f32_bf16<<<nW / 256, 256, 0, stream>>>(Wv, wvb, nW);
  cvt_f32_bf16<<<nW / 256, 256, 0, stream>>>(Wo, wob, nW);

  const float inv4 = 0.17677669529663687f;   // 1024^-0.25
  dim3 gg(kE / 128, kTok / 128);             // (8, 64)
  gemm_bf16<<<gg, 256, 0, stream>>>(xb, wqb, Qb,  nullptr, nullptr, inv4, 0);
  gemm_bf16<<<gg, 256, 0, stream>>>(xb, wkb, Kb,  nullptr, nullptr, inv4, 0);
  gemm_bf16<<<gg, 256, 0, stream>>>(xb, wvb, VTb, nullptr, nullptr, 1.0f, 1);

  attn_fa<<<dim3(kS / 64, kB * kH), 256, 0, stream>>>(Qb, Kb, VTb, Ob);

  gemm_bf16<<<gg, 256, 0, stream>>>(Ob, wob, nullptr, out, bo, 1.0f, 2);
}
